// CosineSimilarity_62053687493017
// MI455X (gfx1250) — matrix operand order repaired and checked by reading, not compiled, not run on hardware
//
#include <hip/hip_runtime.h>
#include <hip/hip_bf16.h>

typedef __attribute__((ext_vector_type(16))) _Float16 v16h;
typedef __attribute__((ext_vector_type(8)))  _Float16 v8h;
typedef __attribute__((ext_vector_type(4)))  _Float16 v4h;
typedef __attribute__((ext_vector_type(8)))  float    v8f;

union V16 { v16h v; v8h h[2]; };

#define TILE_K    128
#define BLK       128
#define LDS_STRIDE 136   // 128 + 8 halves pad -> 272 B row stride, breaks bank striping

// ---------------------------------------------------------------------------
// Prep: one wave per row. Convert f32 -> f16 into ws, compute f32 row norm.
// ---------------------------------------------------------------------------
__global__ __launch_bounds__(256) void cosim_prep_f16(
    const float* __restrict__ in, _Float16* __restrict__ outh,
    float* __restrict__ norms, int nrows) {
  const int wave = threadIdx.x >> 5;
  const int lane = threadIdx.x & 31;
  const int row  = blockIdx.x * 8 + wave;
  if (row >= nrows) return;

  const float4 v = ((const float4*)(in + (size_t)row * TILE_K))[lane];

  v4h h;
  h.x = (_Float16)v.x; h.y = (_Float16)v.y;
  h.z = (_Float16)v.z; h.w = (_Float16)v.w;
  ((v4h*)(outh + (size_t)row * TILE_K))[lane] = h;

  float ss = v.x * v.x + v.y * v.y + v.z * v.z + v.w * v.w;
#pragma unroll
  for (int off = 16; off > 0; off >>= 1)
    ss += __shfl_xor(ss, off, 32);
  if (lane == 0) norms[row] = sqrtf(ss);
}

// ---------------------------------------------------------------------------
// GEMM + normalize. Block = 128x128 output tile, 256 threads (8 waves).
// Wave w computes rows [rowBase, rowBase+16) x 128 cols  (8 WMMA tiles).
// B tile staged in LDS (shared by all waves); A frags stream from global.
// ---------------------------------------------------------------------------
__global__ __launch_bounds__(256) void cosim_wmma_f16(
    const _Float16* __restrict__ Ah, const _Float16* __restrict__ Bh,
    const float* __restrict__ n1g, const float* __restrict__ n2g,
    float* __restrict__ out, int M) {
  __shared__ _Float16 lB[BLK * LDS_STRIDE];

  const int wave    = threadIdx.x >> 5;
  const int lane    = threadIdx.x & 31;
  const int rowBase = blockIdx.y * BLK + wave * 16;
  const int nBase   = blockIdx.x * BLK;

  // ---- cooperative stage of B tile: 128 cols x 128 K halves -> LDS ----
  {
    const int tid = threadIdx.x;
#pragma unroll
    for (int i = 0; i < 16; ++i) {
      const int c    = tid + i * 256;      // 0..4095 chunks of 8 halves
      const int brow = c >> 4;             // column index within tile
      const int kc   = (c & 15) << 3;      // K offset, step 8
      v8h d = *(const v8h*)(Bh + (size_t)(nBase + brow) * TILE_K + kc);
      *(v8h*)(&lB[brow * LDS_STRIDE + kc]) = d;
    }
  }
  __syncthreads();

  v8f acc[8] = {};

  // ISA 16-bit A 16x32 layout: lane&15 = M row; lanes>=16 take K+8 chunk;
  // two 16B chunks per frag at +0 and +16 halves.
  const int aRow  = rowBase + (lane & 15);
  const int aKoff = (lane >> 4) << 3;   // 0 or 8 halves
  // B 32x16 layout (mirror): lane&15 = N col; lanes>=16 take K+8 chunk.
  const int bCol  = lane & 15;
  const int bKoff = (lane >> 4) << 3;   // 0 or 8 halves

#pragma unroll
  for (int k0 = 0; k0 < TILE_K; k0 += 32) {
    V16 a;
    const v8h* pa = (const v8h*)(Ah + (size_t)aRow * TILE_K + k0 + aKoff);
    a.h[0] = pa[0];
    a.h[1] = pa[2];   // +16 halves

#pragma unroll
    for (int t = 0; t < 8; ++t) {
      V16 b;
      const _Float16* pb = &lB[(bCol + t * 16) * LDS_STRIDE + k0 + bKoff];
      b.h[0] = *(const v8h*)pb;
      b.h[1] = *(const v8h*)(pb + 16);
      acc[t] = __builtin_amdgcn_wmma_f32_16x16x32_f16(
          /*neg_a=*/false, a.v, /*neg_b=*/false, b.v,
          /*c_mod=*/(short)0, acc[t], /*reuse_a=*/false, /*reuse_b=*/false);
    }
  }

  // ---- epilogue: D layout — VGPR r holds row (r + 8*(lane>=16)), col = lane&15
  const int hi      = lane >> 4;
  const int ln      = lane & 15;
  const int rowLane = rowBase + hi * 8;

  float na[8];
#pragma unroll
  for (int r = 0; r < 8; ++r) na[r] = n1g[rowLane + r];

#pragma unroll
  for (int t = 0; t < 8; ++t) {
    const int col  = nBase + t * 16 + ln;
    const float nb = n2g[col];
#pragma unroll
    for (int r = 0; r < 8; ++r) {
      const float denom = fmaxf(na[r] * nb, 1e-8f);
      out[(size_t)(rowLane + r) * M + col] = acc[t][r] / denom;
    }
  }
}

// ---------------------------------------------------------------------------
extern "C" void kernel_launch(void* const* d_in, const int* in_sizes, int n_in,
                              void* d_out, int out_size, void* d_ws, size_t ws_size,
                              hipStream_t stream) {
  const float* in1 = (const float*)d_in[0];
  const float* in2 = (const float*)d_in[1];
  float* out = (float*)d_out;

  const int N = in_sizes[0] / TILE_K;   // 8192 rows of input1
  const int M = in_sizes[1] / TILE_K;   // 8192 rows of input2

  _Float16* Ah = (_Float16*)d_ws;
  _Float16* Bh = Ah + (size_t)N * TILE_K;
  float*    n1 = (float*)(Bh + (size_t)M * TILE_K);
  float*    n2 = n1 + N;

  cosim_prep_f16<<<dim3((N + 7) / 8), 256, 0, stream>>>(in1, Ah, n1, N);
  cosim_prep_f16<<<dim3((M + 7) / 8), 256, 0, stream>>>(in2, Bh, n2, M);
  cosim_wmma_f16<<<dim3(M / BLK, N / BLK), 256, 0, stream>>>(Ah, Bh, n1, n2, out, M);
}
